// Model_85796266704925
// MI455X (gfx1250) — matrix operand order repaired and checked by reading, not compiled, not run on hardware
//
#include <hip/hip_runtime.h>
#include <hip/hip_bf16.h>
#include <math.h>

typedef __attribute__((ext_vector_type(16))) _Float16 v16h;
typedef __attribute__((ext_vector_type(8)))  _Float16 v8h;
typedef __attribute__((ext_vector_type(2)))  _Float16 v2h;
typedef __attribute__((ext_vector_type(8)))  float    v8f;

union U16h { v16h v; v8h h[2]; };

#define WMMA_F16(a, b, c) \
  __builtin_amdgcn_wmma_f32_16x16x32_f16(false, (a), false, (b), (short)0, (c), false, false)

// ---- model dimensions (fixed by the reference) ----
static constexpr int Bc  = 2;
static constexpr int Sc  = 4096;
static constexpr int Dc  = 768;
static constexpr int Hc  = 12;
static constexpr int DHc = 64;
static constexpr int NBc = 64;   // sequence blocks of 64
static constexpr int Lc  = 4;
static constexpr int FFc = 3072;
static constexpr int Mc  = Bc * Sc;   // 8192 tokens

// ----------------------------------------------------------------------------
// helpers
// ----------------------------------------------------------------------------
__device__ __forceinline__ float gelu_f(float x) {
  float x3 = x * x * x;
  return 0.5f * x * (1.0f + tanhf(0.7978845608f * (x + 0.044715f * x3)));
}

__device__ __forceinline__ float blk_sum256(float v, float* sh) {
  int t = threadIdx.x;
  sh[t] = v; __syncthreads();
  for (int o = 128; o > 0; o >>= 1) { if (t < o) sh[t] += sh[t + o]; __syncthreads(); }
  float r = sh[0]; __syncthreads();
  return r;
}
__device__ __forceinline__ float blk_max256(float v, float* sh) {
  int t = threadIdx.x;
  sh[t] = v; __syncthreads();
  for (int o = 128; o > 0; o >>= 1) { if (t < o) sh[t] = fmaxf(sh[t], sh[t + o]); __syncthreads(); }
  float r = sh[0]; __syncthreads();
  return r;
}

// ----------------------------------------------------------------------------
// 1) embedding gather + LayerNorm -> fp32 residual stream + f16 GEMM input.
// ----------------------------------------------------------------------------
__global__ void embed_ln_kernel(const int* __restrict__ ids,
                                const float* __restrict__ wemb,
                                const float* __restrict__ pemb,
                                const float* __restrict__ g,
                                const float* __restrict__ bta,
                                float* __restrict__ out,
                                _Float16* __restrict__ outh) {
  __shared__ float sh[256];
  int tok = blockIdx.x;
  int s   = tok % Sc;
  int id  = ids[tok];
  const float* we = wemb + (size_t)id * Dc;
  const float* pe = pemb + (size_t)s  * Dc;
  float r[3], lsum = 0.f, lsq = 0.f;
  #pragma unroll
  for (int i = 0; i < 3; ++i) {
    int d = threadIdx.x + i * 256;
    float v = we[d] + pe[d];
    r[i] = v; lsum += v; lsq += v * v;
  }
  float sum = blk_sum256(lsum, sh);
  float sq  = blk_sum256(lsq,  sh);
  float mean = sum * (1.0f / Dc);
  float var  = sq * (1.0f / Dc) - mean * mean;
  float rstd = rsqrtf(var + 1e-12f);
  float*    op  = out  + (size_t)tok * Dc;
  _Float16* oph = outh + (size_t)tok * Dc;
  #pragma unroll
  for (int i = 0; i < 3; ++i) {
    int d = threadIdx.x + i * 256;
    float y = (r[i] - mean) * rstd * g[d] + bta[d];
    op[d]  = y;
    oph[d] = (_Float16)y;
  }
}

// ----------------------------------------------------------------------------
// 2) residual + LayerNorm: out = LN(base + resid); optional f16 copy.
// ----------------------------------------------------------------------------
__global__ void add_ln_kernel(const float* __restrict__ base,
                              const float* __restrict__ resid,
                              const float* __restrict__ g,
                              const float* __restrict__ bta,
                              float* __restrict__ out,
                              _Float16* __restrict__ outh) {
  __shared__ float sh[256];
  int tok = blockIdx.x;
  const float* bp = base + (size_t)tok * Dc;
  const float* rp = resid ? resid + (size_t)tok * Dc : nullptr;
  float r[3], lsum = 0.f, lsq = 0.f;
  #pragma unroll
  for (int i = 0; i < 3; ++i) {
    int d = threadIdx.x + i * 256;
    float v = bp[d] + (rp ? rp[d] : 0.0f);
    r[i] = v; lsum += v; lsq += v * v;
  }
  float sum = blk_sum256(lsum, sh);
  float sq  = blk_sum256(lsq,  sh);
  float mean = sum * (1.0f / Dc);
  float var  = sq * (1.0f / Dc) - mean * mean;
  float rstd = rsqrtf(var + 1e-12f);
  float*    op  = out + (size_t)tok * Dc;
  _Float16* oph = outh ? outh + (size_t)tok * Dc : nullptr;
  #pragma unroll
  for (int i = 0; i < 3; ++i) {
    int d = threadIdx.x + i * 256;
    float y = (r[i] - mean) * rstd * g[d] + bta[d];
    op[d] = y;
    if (oph) oph[d] = (_Float16)y;
  }
}

// ----------------------------------------------------------------------------
// 3) GEMM: C = act(A(MxK,f16) * W(KxN,f32) + bias).  f16 WMMA, f32 accum.
//    block = 256 (8 waves); block tile 128x64; wave owns a 16x64 strip.
//    Weight k-tiles (32x64) ping-pong through LDS (transposed f16 WhT[n][k]).
//    A fragments are pure wide f16 loads (no conversion in the k-loop).
//    OUTMODE: 0 = write fp32 C; 2 = write f16 Ch only (WMMA-consumer output).
// ----------------------------------------------------------------------------
template <int ACT, int OUTMODE>
__global__ void gemm_wmma_kernel(const _Float16* __restrict__ A,
                                 const float* __restrict__ W,
                                 const float* __restrict__ bias,
                                 float* __restrict__ C,
                                 _Float16* __restrict__ Ch,
                                 int M, int N, int K) {
  __shared__ _Float16 WhT[2][64][32];   // ping-pong [n][k], k contiguous (8 KB)

  const int wave = threadIdx.x >> 5;          // 0..7
  const int lane = threadIdx.x & 31;
  const int ln   = lane & 15;
  const int hl   = lane >> 4;
  const int m0   = blockIdx.y * 128 + wave * 16;
  const int n0   = blockIdx.x * 64;
  const int sn   = threadIdx.x & 63;          // staging n
  const int st6  = (threadIdx.x >> 6) & 3;    // staging k group

  v8f acc[4];
  #pragma unroll
  for (int nt = 0; nt < 4; ++nt)
    #pragma unroll
    for (int v = 0; v < 8; ++v) acc[nt][v] = 0.0f;

  const _Float16* arow  = A + (size_t)(m0 + ln) * K;
  const float*    wbase = W + n0 + sn;

  // prologue: stage k-tile 0 into buffer 0
  {
    const float* wp = wbase + (size_t)(2 * st6) * N;
    #pragma unroll
    for (int i = 0; i < 4; ++i) {
      v2h pk;
      pk[0] = (_Float16)wp[0];
      pk[1] = (_Float16)wp[N];
      *(v2h*)&WhT[0][sn][8 * i + 2 * st6] = pk;
      wp += (size_t)8 * N;
    }
  }

  int cur = 0;
  for (int k0 = 0; k0 < K; k0 += 32) {
    __syncthreads();   // buffer `cur` staged & previous reads done

    // A fragment: two contiguous 16B f16 loads (issued before staging loads)
    U16h af;
    af.h[0] = *(const v8h*)(arow + k0 + 16 * hl);
    af.h[1] = *(const v8h*)(arow + k0 + 16 * hl + 8);

    // issue next k-tile's weight loads (no wait yet)
    float sva[4], svb[4];
    const bool more = (k0 + 32 < K);
    if (more) {
      const float* wp = wbase + (size_t)(k0 + 32 + 2 * st6) * N;
      #pragma unroll
      for (int i = 0; i < 4; ++i) { sva[i] = wp[0]; svb[i] = wp[N]; wp += (size_t)8 * N; }
      __builtin_prefetch(arow + k0 + 64 + 8 * hl, 0, 1);
    }

    // compute from current buffer (B frag = one contiguous 32B LDS read)
    #pragma unroll
    for (int nt = 0; nt < 4; ++nt) {
      v16h bf = *(const v16h*)&WhT[cur][nt * 16 + ln][16 * hl];
      acc[nt] = WMMA_F16(af.v, bf, acc[nt]);
    }

    // convert + store staged tile (loadcnt wait lands here, after the WMMAs)
    if (more) {
      #pragma unroll
      for (int i = 0; i < 4; ++i) {
        v2h pk;
        pk[0] = (_Float16)sva[i];
        pk[1] = (_Float16)svb[i];
        *(v2h*)&WhT[cur ^ 1][sn][8 * i + 2 * st6] = pk;
      }
    }
    cur ^= 1;
  }

  // ---- store: C layout row = v + 8*hl, col = lane&15 (coalesced over lanes) --
  #pragma unroll
  for (int nt = 0; nt < 4; ++nt) {
    int n = n0 + nt * 16 + ln;
    float bv = bias[n];
    #pragma unroll
    for (int v = 0; v < 8; ++v) {
      float x = acc[nt][v] + bv;
      if (ACT == 1) x = gelu_f(x);
      size_t off = (size_t)(m0 + v + 8 * hl) * N + n;
      if (OUTMODE == 0) C[off]  = x;
      if (OUTMODE == 2) Ch[off] = (_Float16)x;
    }
  }
}

// ----------------------------------------------------------------------------
// 4) Flash attention with WMMA over f16 Q/K/V (BigBird sparse / global rows).
//    block = 128 (4 waves); wave owns 16 query rows of a 64-row query block.
//    V-chunk staging loads overlap the softmax VALU work; output written f16.
// ----------------------------------------------------------------------------
template <bool GLOBALM>
__global__ void attn_kernel(const _Float16* __restrict__ Qm,
                            const _Float16* __restrict__ Km,
                            const _Float16* __restrict__ Vm,
                            const int* __restrict__ randb,
                            const int* __restrict__ amask,
                            _Float16* __restrict__ Om) {
  const int h    = blockIdx.y;
  const int b    = blockIdx.z;
  const int iblk = GLOBALM ? (blockIdx.x == 0 ? 0 : NBc - 1) : blockIdx.x;
  const int wave = threadIdx.x >> 5;
  const int lane = threadIdx.x & 31;
  const int ln   = lane & 15;
  const int hl   = lane >> 4;
  const float scale = 0.125f;                 // 1/sqrt(64)

  __shared__ _Float16 plds[4][16][64];   // per-wave probability staging (8 KB)
  __shared__ _Float16 VtT[64][64];       // V chunk transposed [d][key]   (8 KB)

  // preload Q fragments (A layout: 16 contiguous dims at 16*hl per lane)
  U16h qa0, qa1;
  {
    int srow = iblk * 64 + wave * 16 + ln;
    const _Float16* qp = Qm + ((size_t)(b * Sc + srow)) * Dc + h * DHc;
    qa0.h[0] = *(const v8h*)(qp + 16 * hl);
    qa0.h[1] = *(const v8h*)(qp + 16 * hl + 8);
    qa1.h[0] = *(const v8h*)(qp + 32 + 16 * hl);
    qa1.h[1] = *(const v8h*)(qp + 32 + 16 * hl + 8);
  }

  v8f acc[4];
  #pragma unroll
  for (int nt = 0; nt < 4; ++nt)
    #pragma unroll
    for (int v = 0; v < 8; ++v) acc[nt][v] = 0.0f;
  float mrun[8], lrun[8];
  #pragma unroll
  for (int v = 0; v < 8; ++v) { mrun[v] = -1e30f; lrun[v] = 0.0f; }

  int kblist[8];
  int nch;
  if (GLOBALM) {
    nch = NBc;
  } else {
    int i = iblk;
    kblist[0] = 0;
    kblist[1] = (i - 1 < 0) ? 0 : i - 1;
    kblist[2] = i;
    kblist[3] = (i + 1 > NBc - 1) ? NBc - 1 : i + 1;
    kblist[4] = NBc - 1;
    kblist[5] = randb[i * 3 + 0];
    kblist[6] = randb[i * 3 + 1];
    kblist[7] = randb[i * 3 + 2];
    nch = 8;
  }

  for (int c = 0; c < nch; ++c) {
    int kblk;
    if (GLOBALM) {
      kblk = c;
    } else {
      kblk = kblist[c];
      bool dup = false;                       // uniform across the block
      for (int j = 0; j < c; ++j) dup |= (kblist[j] == kblk);
      if (dup) continue;                      // -1e9 dup bias == exact zero weight
    }

    // prefetch next chunk's K rows
    if (c + 1 < nch) {
      int nk = GLOBALM ? (c + 1) : kblist[c + 1];
      const _Float16* pf = Km + ((size_t)(b * Sc + nk * 64 + lane)) * Dc + h * DHc;
      __builtin_prefetch(pf, 0, 1);
      __builtin_prefetch(pf + (size_t)32 * Dc, 0, 1);
    }

    // ---- scores: S = Q * K^T * scale + mask_bias (pure wide f16 loads) ----
    float st[4][8];
    #pragma unroll
    for (int nt = 0; nt < 4; ++nt) {
      int key = kblk * 64 + nt * 16 + ln;
      float mb = (amask[b * Sc + key] != 0) ? 0.0f : -1e9f;
      const _Float16* kp = Km + ((size_t)(b * Sc + key)) * Dc + h * DHc;
      U16h bk0, bk1;                 // B layout: 16 contiguous dims at 16*hl
      bk0.h[0] = *(const v8h*)(kp + 16 * hl);
      bk0.h[1] = *(const v8h*)(kp + 16 * hl + 8);
      bk1.h[0] = *(const v8h*)(kp + 32 + 16 * hl);
      bk1.h[1] = *(const v8h*)(kp + 32 + 16 * hl + 8);
      v8f sv;
      #pragma unroll
      for (int v = 0; v < 8; ++v) sv[v] = 0.0f;
      sv = WMMA_F16(qa0.v, bk0.v, sv);
      sv = WMMA_F16(qa1.v, bk1.v, sv);
      #pragma unroll
      for (int v = 0; v < 8; ++v) st[nt][v] = sv[v] * scale + mb;
    }

    // ---- issue V chunk staging loads (overlap with softmax below) ----
    // thread -> key = kblk*64 + 4*i + wave, dims (2*lane, 2*lane+1)
    v2h vreg[16];
    {
      const _Float16* vp = Vm + ((size_t)(b * Sc + kblk * 64 + wave)) * Dc + h * DHc + 2 * lane;
      #pragma unroll
      for (int i = 0; i < 16; ++i) {
        vreg[i] = *(const v2h*)vp;
        vp += (size_t)4 * Dc;
      }
    }

    // ---- online softmax update (per row v, across 16 lanes x 4 tiles) ----
    #pragma unroll
    for (int v = 0; v < 8; ++v) {
      float tm = fmaxf(fmaxf(st[0][v], st[1][v]), fmaxf(st[2][v], st[3][v]));
      #pragma unroll
      for (int o = 1; o < 16; o <<= 1) tm = fmaxf(tm, __shfl_xor(tm, o, 32));
      float mnew = fmaxf(mrun[v], tm);
      float corr = __expf(mrun[v] - mnew);
      mrun[v] = mnew;
      lrun[v] *= corr;
      #pragma unroll
      for (int nt = 0; nt < 4; ++nt) acc[nt][v] *= corr;
      float ps = 0.0f;
      #pragma unroll
      for (int nt = 0; nt < 4; ++nt) {
        float p = __expf(st[nt][v] - mnew);
        st[nt][v] = p;
        ps += p;
      }
      #pragma unroll
      for (int o = 1; o < 16; o <<= 1) ps += __shfl_xor(ps, o, 32);
      lrun[v] += ps;
    }

    // ---- stage P (C-layout) to LDS as f16 ----
    #pragma unroll
    for (int nt = 0; nt < 4; ++nt)
      #pragma unroll
      for (int v = 0; v < 8; ++v)
        plds[wave][v + 8 * hl][nt * 16 + ln] = (_Float16)st[nt][v];

    // ---- store staged V chunk transposed (loadcnt wait lands here) ----
    #pragma unroll
    for (int i = 0; i < 16; ++i) {
      int key = 4 * i + wave;
      VtT[2 * lane][key]     = vreg[i][0];
      VtT[2 * lane + 1][key] = vreg[i][1];
    }
    __syncthreads();   // plds + VtT visible

    // P as A fragments: contiguous 16B (v8h) LDS reads
    U16h u0, u1;
    u0.h[0] = *(const v8h*)&plds[wave][ln][16 * hl];
    u0.h[1] = *(const v8h*)&plds[wave][ln][16 * hl + 8];
    u1.h[0] = *(const v8h*)&plds[wave][ln][32 + 16 * hl];
    u1.h[1] = *(const v8h*)&plds[wave][ln][32 + 16 * hl + 8];

    // ---- O += P * V  (V fragments = contiguous 32B LDS reads) ----
    #pragma unroll
    for (int nt = 0; nt < 4; ++nt) {
      int dcol = nt * 16 + ln;
      v16h bv0 = *(const v16h*)&VtT[dcol][16 * hl];
      v16h bv1 = *(const v16h*)&VtT[dcol][32 + 16 * hl];
      acc[nt] = WMMA_F16(u0.v, bv0, acc[nt]);
      acc[nt] = WMMA_F16(u1.v, bv1, acc[nt]);
    }
    __syncthreads();   // protect plds/VtT before next chunk overwrites
  }

  // ---- normalize and write f16 [b, s, h*64 + d] ----
  #pragma unroll
  for (int v = 0; v < 8; ++v) {
    int srow = iblk * 64 + wave * 16 + v + 8 * hl;
    float inv = (lrun[v] > 0.0f) ? 1.0f / lrun[v] : 0.0f;
    _Float16* op = Om + ((size_t)(b * Sc + srow)) * Dc + h * DHc;
    #pragma unroll
    for (int nt = 0; nt < 4; ++nt) op[nt * 16 + ln] = (_Float16)(acc[nt][v] * inv);
  }
}

// ----------------------------------------------------------------------------
// 5) QA head: logits = h @ qa_w + qa_b, zeroed where question_mask.
// ----------------------------------------------------------------------------
__global__ void qa_head_kernel(const float* __restrict__ Hx,
                               const float* __restrict__ qaw,
                               const float* __restrict__ qab,
                               const int* __restrict__ qmask,
                               float* __restrict__ slog,
                               float* __restrict__ elog) {
  int t = blockIdx.x * blockDim.x + threadIdx.x;
  if (t >= Mc) return;
  const float* hp = Hx + (size_t)t * Dc;
  float a0 = qab[0], a1 = qab[1];
  for (int d = 0; d < Dc; ++d) {
    float v = hp[d];
    a0 += v * qaw[2 * d];
    a1 += v * qaw[2 * d + 1];
  }
  if (qmask[t] != 0) { a0 = 0.0f; a1 = 0.0f; }
  slog[t] = a0;
  elog[t] = a1;
}

// ----------------------------------------------------------------------------
// 6) CE-with-ignore loss.  Single block of 256.
// ----------------------------------------------------------------------------
__global__ void loss_kernel(const float* __restrict__ slog,
                            const float* __restrict__ elog,
                            const int* __restrict__ spos,
                            const int* __restrict__ epos,
                            float* __restrict__ out) {
  __shared__ float sh[256];
  float ce[2];
  for (int which = 0; which < 2; ++which) {
    const float* lg  = which ? elog : slog;
    const int*   lab = which ? epos : spos;
    float nll_sum = 0.0f, w_sum = 0.0f;
    for (int b = 0; b < Bc; ++b) {
      const float* row = lg + (size_t)b * Sc;
      float mx = -1e30f;
      for (int i = threadIdx.x; i < Sc; i += 256) mx = fmaxf(mx, row[i]);
      mx = blk_max256(mx, sh);
      float se = 0.0f;
      for (int i = threadIdx.x; i < Sc; i += 256) se += __expf(row[i] - mx);
      se = blk_sum256(se, sh);
      int L = lab[b];
      if (L >= 0) {
        float lp = row[L] - mx - logf(se);
        nll_sum += -lp;
        w_sum   += 1.0f;
      }
    }
    ce[which] = nll_sum / fmaxf(w_sum, 1.0f);
  }
  if (threadIdx.x == 0) out[0] = 0.5f * (ce[0] + ce[1]);
}

// ----------------------------------------------------------------------------
// launcher
// ----------------------------------------------------------------------------
extern "C" void kernel_launch(void* const* d_in, const int* in_sizes, int n_in,
                              void* d_out, int out_size, void* d_ws, size_t ws_size,
                              hipStream_t stream) {
  const int*   input_ids  = (const int*)  d_in[0];
  const int*   attn_mask  = (const int*)  d_in[1];
  const int*   qmask      = (const int*)  d_in[2];
  const int*   spos       = (const int*)  d_in[3];
  const int*   epos       = (const int*)  d_in[4];
  const int*   rand_blk   = (const int*)  d_in[5];
  const float* word_emb   = (const float*)d_in[6];
  const float* pos_emb    = (const float*)d_in[7];
  const float* emb_ln_s   = (const float*)d_in[8];
  const float* emb_ln_b   = (const float*)d_in[9];
  const float* Wq = (const float*)d_in[10]; const float* bq = (const float*)d_in[11];
  const float* Wk = (const float*)d_in[12]; const float* bk = (const float*)d_in[13];
  const float* Wv = (const float*)d_in[14]; const float* bv = (const float*)d_in[15];
  const float* Wo = (const float*)d_in[16]; const float* bo = (const float*)d_in[17];
  const float* ln1s = (const float*)d_in[18]; const float* ln1b = (const float*)d_in[19];
  const float* Wff1 = (const float*)d_in[20]; const float* bff1 = (const float*)d_in[21];
  const float* Wff2 = (const float*)d_in[22]; const float* bff2 = (const float*)d_in[23];
  const float* ln2s = (const float*)d_in[24]; const float* ln2b = (const float*)d_in[25];
  const float* d1w = (const float*)d_in[26]; const float* d1b = (const float*)d_in[27];
  const float* d2w = (const float*)d_in[28]; const float* d2b = (const float*)d_in[29];
  const float* hlns = (const float*)d_in[30]; const float* hlnb = (const float*)d_in[31];
  const float* qaw = (const float*)d_in[32]; const float* qab = (const float*)d_in[33];

  float* outp = (float*)d_out;   // [loss(1) | start_logits(8192) | end_logits(8192)]
  float* slog = outp + 1;
  float* elog = outp + 1 + Mc;

  // workspace: 2 fp32 activation buffers + f16 activation stream (~165 MB)
  const size_t MD = (size_t)Mc * Dc;
  float*    X  = (float*)d_ws;        // fp32 residual stream
  float*    Y  = X + MD;              // fp32 GEMM output (pre-LN)
  _Float16* Xh = (_Float16*)(Y + MD); // f16 copy of X (GEMM input)
  _Float16* Qh = Xh + MD;
  _Float16* Kh = Qh + MD;
  _Float16* Vh = Kh + MD;
  _Float16* Ah = Vh + MD;             // attention output (f16)
  _Float16* Fh = Ah + MD;             // FFN hidden (M x 3072, f16)

  const dim3 blk128(128), blk256(256);
  const dim3 gD (Dc  / 64, Mc / 128);
  const dim3 gFF(FFc / 64, Mc / 128);

  embed_ln_kernel<<<Mc, blk256, 0, stream>>>(input_ids, word_emb, pos_emb,
                                             emb_ln_s, emb_ln_b, X, Xh);

  for (int l = 0; l < Lc; ++l) {
    const size_t wDD = (size_t)l * Dc * Dc;
    const size_t wDF = (size_t)l * Dc * FFc;
    gemm_wmma_kernel<0,2><<<gD, blk256, 0, stream>>>(Xh, Wq + wDD, bq + l * Dc,
                                                     nullptr, Qh, Mc, Dc, Dc);
    gemm_wmma_kernel<0,2><<<gD, blk256, 0, stream>>>(Xh, Wk + wDD, bk + l * Dc,
                                                     nullptr, Kh, Mc, Dc, Dc);
    gemm_wmma_kernel<0,2><<<gD, blk256, 0, stream>>>(Xh, Wv + wDD, bv + l * Dc,
                                                     nullptr, Vh, Mc, Dc, Dc);

    attn_kernel<false><<<dim3(NBc, Hc, Bc), blk128, 0, stream>>>(Qh, Kh, Vh, rand_blk,
                                                                 attn_mask, Ah);
    attn_kernel<true ><<<dim3(2,   Hc, Bc), blk128, 0, stream>>>(Qh, Kh, Vh, rand_blk,
                                                                 attn_mask, Ah);

    gemm_wmma_kernel<0,0><<<gD, blk256, 0, stream>>>(Ah, Wo + wDD, bo + l * Dc,
                                                     Y, nullptr, Mc, Dc, Dc);
    add_ln_kernel<<<Mc, blk256, 0, stream>>>(X, Y, ln1s + l * Dc, ln1b + l * Dc, X, Xh);

    gemm_wmma_kernel<1,2><<<gFF, blk256, 0, stream>>>(Xh, Wff1 + wDF, bff1 + l * FFc,
                                                      nullptr, Fh, Mc, FFc, Dc);
    gemm_wmma_kernel<0,0><<<gD,  blk256, 0, stream>>>(Fh, Wff2 + wDF, bff2 + l * Dc,
                                                      Y, nullptr, Mc, Dc, FFc);
    add_ln_kernel<<<Mc, blk256, 0, stream>>>(X, Y, ln2s + l * Dc, ln2b + l * Dc, X, Xh);
  }

  gemm_wmma_kernel<1,2><<<gFF, blk256, 0, stream>>>(Xh, d1w, d1b, nullptr, Fh, Mc, FFc, Dc);
  gemm_wmma_kernel<0,0><<<gD,  blk256, 0, stream>>>(Fh, d2w, d2b, Y, nullptr, Mc, Dc, FFc);
  add_ln_kernel<<<Mc, blk256, 0, stream>>>(Y, nullptr, hlns, hlnb, X, nullptr);

  qa_head_kernel<<<Mc / 256, blk256, 0, stream>>>(X, qaw, qab, qmask, slog, elog);
  loss_kernel<<<1, blk256, 0, stream>>>(slog, elog, spos, epos, outp);
}
